// _EmformerAttention_32950989094997
// MI455X (gfx1250) — matrix operand order repaired and checked by reading, not compiled, not run on hardware
//
#include <hip/hip_runtime.h>
#include <math.h>

// ---------------- types ----------------
typedef __attribute__((ext_vector_type(16))) __bf16 v16bf;
typedef __attribute__((ext_vector_type(8)))  __bf16 v8bf;
typedef __attribute__((ext_vector_type(8)))  float  v8f;

#define WMMA_BF16(a, b, c) \
  __builtin_amdgcn_wmma_f32_16x16x32_bf16(false, (a), false, (b), (short)0, (c), false, false)

// ---------------- problem dims ----------------
constexpr int U_ = 512, R_ = 64, S_ = 1, M_ = 8, L_ = 128, B_ = 32, D_ = 1024, H_ = 8, DH = 128;
constexpr int T_  = R_ + U_ + S_;        // 577 query positions
constexpr int KV_ = M_ + R_ + U_ + L_;   // 712 kv positions
constexpr int QROWS  = T_ * B_;          // 18464
constexpr int KVROWS = (M_ + R_ + U_) * B_; // 18688
constexpr int VSTRIDE = 736;             // 712 padded to 23*32 (chunk-aligned)
constexpr int KSTEPS = D_ / 32;          // 32 K-steps in projections
constexpr float SCALING = 0.08838834764831845f; // 128^-0.5
constexpr float NEG_INF_ = -100000000.0f;

__device__ inline v16bf cat8(v8bf lo, v8bf hi) {
  return __builtin_shufflevector(lo, hi, 0,1,2,3,4,5,6,7,8,9,10,11,12,13,14,15);
}

__device__ inline v16bf cvt16(const float4& x0, const float4& x1,
                              const float4& y0, const float4& y1) {
  float tf[16] = {x0.x,x0.y,x0.z,x0.w, x1.x,x1.y,x1.z,x1.w,
                  y0.x,y0.y,y0.z,y0.w, y1.x,y1.y,y1.z,y1.w};
  v16bf a;
#pragma unroll
  for (int i = 0; i < 16; ++i) a[i] = (__bf16)tf[i];
  return a;
}

// ---------------- fp32 -> bf16 bulk convert ----------------
__global__ void cvt_f32_bf16(const float* __restrict__ src, __bf16* __restrict__ dst, int n) {
  int i = blockIdx.x * 256 + threadIdx.x;
  if (i < n) dst[i] = (__bf16)src[i];
}

// =====================================================================
// Projection GEMMs: block = 8 waves; each wave owns 32 M-rows x 64 N.
// B panel (64n x 32k bf16) double-buffered in LDS (compile-time buffer
// pointers, pair-unrolled loop); A raw data ping-pongs between two
// register arrays -> no rotation copies, in-place WMMA accumulation.
// =====================================================================

// ---------------- Q projection -> q_bf [bh][T][128] (scaled)
__global__ __launch_bounds__(256) void gemm_q_kernel(
    const float* __restrict__ utt, const float* __restrict__ rc, const float* __restrict__ summ,
    const __bf16* __restrict__ Wq, const float* __restrict__ bq,
    __bf16* __restrict__ qbf)
{
  __shared__ __attribute__((aligned(32))) __bf16 Bs[2][64 * 32];
  const int tid  = threadIdx.x;
  const int lane = tid & 31, wave = tid >> 5;
  const int lmod = lane & 15, lhalf = lane >> 4;
  const int m0 = blockIdx.y * 256 + wave * 32;
  const int n0 = blockIdx.x * 64;

  const int brow = tid >> 2, bq4 = tid & 3;
  const __bf16* bsrc = Wq + (size_t)(n0 + brow) * D_ + bq4 * 8;
  const int boff = brow * 32 + ((bq4 & 1) * 2 + (bq4 >> 1)) * 8;

  auto qrow_ptr = [&](int row) -> const float* {
    if (row >= QROWS) row = QROWS - 1;
    const int t = row >> 5, b = row & 31;
    if (t < R_)           return rc   + ((size_t)t * B_ + b) * D_;
    if (t < R_ + U_)      return utt  + ((size_t)(t - R_) * B_ + b) * D_;
    return summ + (size_t)b * D_;
  };
  const float* arp0 = qrow_ptr(m0 + lmod) + lhalf * 8;
  const float* arp1 = qrow_ptr(m0 + 16 + lmod) + lhalf * 8;

  // prologue
  *(v8bf*)(&Bs[0][boff]) = *(const v8bf*)(bsrc);
  float4 arA[8], arB[8];
#pragma unroll
  for (int i = 0; i < 2; ++i) {
    arA[4*i+0] = ((const float4*)((i ? arp1 : arp0)))[0];
    arA[4*i+1] = ((const float4*)((i ? arp1 : arp0)))[1];
    arA[4*i+2] = ((const float4*)((i ? arp1 : arp0) + 16))[0];
    arA[4*i+3] = ((const float4*)((i ? arp1 : arp0) + 16))[1];
  }
  __syncthreads();

  v8f acc[8];
#pragma unroll
  for (int j = 0; j < 8; ++j) acc[j] = v8f{};

  auto step = [&](int s, __bf16* BsCur, __bf16* BsNext, float4* arCur, float4* arNext) {
    const bool more = (s + 1) < KSTEPS;
    v8bf brn{};
    if (more) {
      const int kn = (s + 1) * 32;
      brn = *(const v8bf*)(bsrc + kn);
      arNext[0] = ((const float4*)(arp0 + kn))[0];
      arNext[1] = ((const float4*)(arp0 + kn))[1];
      arNext[2] = ((const float4*)(arp0 + kn + 16))[0];
      arNext[3] = ((const float4*)(arp0 + kn + 16))[1];
      arNext[4] = ((const float4*)(arp1 + kn))[0];
      arNext[5] = ((const float4*)(arp1 + kn))[1];
      arNext[6] = ((const float4*)(arp1 + kn + 16))[0];
      arNext[7] = ((const float4*)(arp1 + kn + 16))[1];
    }
    v16bf aLo = cvt16(arCur[0], arCur[1], arCur[2], arCur[3]);
    v16bf aHi = cvt16(arCur[4], arCur[5], arCur[6], arCur[7]);
    const __bf16* bb = BsCur + lmod * 32 + lhalf * 16;
    v16bf b0 = *(const v16bf*)(bb +  0 * 32);
    v16bf b1 = *(const v16bf*)(bb + 16 * 32);
    v16bf b2 = *(const v16bf*)(bb + 32 * 32);
    v16bf b3 = *(const v16bf*)(bb + 48 * 32);
    acc[0] = WMMA_BF16(aLo, b0, acc[0]);
    acc[1] = WMMA_BF16(aLo, b1, acc[1]);
    acc[2] = WMMA_BF16(aLo, b2, acc[2]);
    acc[3] = WMMA_BF16(aLo, b3, acc[3]);
    acc[4] = WMMA_BF16(aHi, b0, acc[4]);
    acc[5] = WMMA_BF16(aHi, b1, acc[5]);
    acc[6] = WMMA_BF16(aHi, b2, acc[6]);
    acc[7] = WMMA_BF16(aHi, b3, acc[7]);
    if (more) {
      *(v8bf*)(BsNext + boff) = brn;
      __syncthreads();
    }
  };

  for (int s2 = 0; s2 < KSTEPS / 2; ++s2) {
    step(2 * s2,     &Bs[0][0], &Bs[1][0], arA, arB);
    step(2 * s2 + 1, &Bs[1][0], &Bs[0][0], arB, arA);
  }

  const float bias[4] = { bq[n0 + lmod], bq[n0 + 16 + lmod], bq[n0 + 32 + lmod], bq[n0 + 48 + lmod] };
#pragma unroll
  for (int hi = 0; hi < 2; ++hi)
#pragma unroll
  for (int r = 0; r < 8; ++r) {
    const int m = m0 + 16 * hi + r + 8 * lhalf;
    if (m >= QROWS) continue;
    const int tt = m >> 5, bb = m & 31;
#pragma unroll
    for (int j = 0; j < 4; ++j) {
      const int n = n0 + 16 * j + lmod;
      const int h = n >> 7, dd = n & 127;
      float v = acc[4 * hi + j][r] + bias[j];
      qbf[((size_t)(bb * H_ + h) * T_ + tt) * DH + dd] = (__bf16)(v * SCALING);
    }
  }
}

// ---------------- KV projection -> k_bf, vT_bf + fp32 key/value outputs
__global__ __launch_bounds__(256) void gemm_kv_kernel(
    const float* __restrict__ mems, const float* __restrict__ rc, const float* __restrict__ utt,
    const __bf16* __restrict__ Wkv, const float* __restrict__ bkv,
    __bf16* __restrict__ kbf, __bf16* __restrict__ vT,
    float* __restrict__ out_key, float* __restrict__ out_val)
{
  __shared__ __attribute__((aligned(32))) __bf16 Bs[2][64 * 32];
  const int tid  = threadIdx.x;
  const int lane = tid & 31, wave = tid >> 5;
  const int lmod = lane & 15, lhalf = lane >> 4;
  const int m0 = blockIdx.y * 256 + wave * 32;
  const int n0 = blockIdx.x * 64;

  const int brow = tid >> 2, bq4 = tid & 3;
  const __bf16* bsrc = Wkv + (size_t)(n0 + brow) * D_ + bq4 * 8;
  const int boff = brow * 32 + ((bq4 & 1) * 2 + (bq4 >> 1)) * 8;

  auto kvrow_ptr = [&](int row) -> const float* {
    if (row >= KVROWS) row = KVROWS - 1;
    const int t = row >> 5, b = row & 31;
    if (t < M_)           return mems + ((size_t)t * B_ + b) * D_;
    if (t < M_ + R_)      return rc   + ((size_t)(t - M_) * B_ + b) * D_;
    return utt + ((size_t)(t - M_ - R_) * B_ + b) * D_;
  };
  const float* arp0 = kvrow_ptr(m0 + lmod) + lhalf * 8;
  const float* arp1 = kvrow_ptr(m0 + 16 + lmod) + lhalf * 8;

  *(v8bf*)(&Bs[0][boff]) = *(const v8bf*)(bsrc);
  float4 arA[8], arB[8];
#pragma unroll
  for (int i = 0; i < 2; ++i) {
    arA[4*i+0] = ((const float4*)((i ? arp1 : arp0)))[0];
    arA[4*i+1] = ((const float4*)((i ? arp1 : arp0)))[1];
    arA[4*i+2] = ((const float4*)((i ? arp1 : arp0) + 16))[0];
    arA[4*i+3] = ((const float4*)((i ? arp1 : arp0) + 16))[1];
  }
  __syncthreads();

  v8f acc[8];
#pragma unroll
  for (int j = 0; j < 8; ++j) acc[j] = v8f{};

  auto step = [&](int s, __bf16* BsCur, __bf16* BsNext, float4* arCur, float4* arNext) {
    const bool more = (s + 1) < KSTEPS;
    v8bf brn{};
    if (more) {
      const int kn = (s + 1) * 32;
      brn = *(const v8bf*)(bsrc + kn);
      arNext[0] = ((const float4*)(arp0 + kn))[0];
      arNext[1] = ((const float4*)(arp0 + kn))[1];
      arNext[2] = ((const float4*)(arp0 + kn + 16))[0];
      arNext[3] = ((const float4*)(arp0 + kn + 16))[1];
      arNext[4] = ((const float4*)(arp1 + kn))[0];
      arNext[5] = ((const float4*)(arp1 + kn))[1];
      arNext[6] = ((const float4*)(arp1 + kn + 16))[0];
      arNext[7] = ((const float4*)(arp1 + kn + 16))[1];
    }
    v16bf aLo = cvt16(arCur[0], arCur[1], arCur[2], arCur[3]);
    v16bf aHi = cvt16(arCur[4], arCur[5], arCur[6], arCur[7]);
    const __bf16* bb = BsCur + lmod * 32 + lhalf * 16;
    v16bf b0 = *(const v16bf*)(bb +  0 * 32);
    v16bf b1 = *(const v16bf*)(bb + 16 * 32);
    v16bf b2 = *(const v16bf*)(bb + 32 * 32);
    v16bf b3 = *(const v16bf*)(bb + 48 * 32);
    acc[0] = WMMA_BF16(aLo, b0, acc[0]);
    acc[1] = WMMA_BF16(aLo, b1, acc[1]);
    acc[2] = WMMA_BF16(aLo, b2, acc[2]);
    acc[3] = WMMA_BF16(aLo, b3, acc[3]);
    acc[4] = WMMA_BF16(aHi, b0, acc[4]);
    acc[5] = WMMA_BF16(aHi, b1, acc[5]);
    acc[6] = WMMA_BF16(aHi, b2, acc[6]);
    acc[7] = WMMA_BF16(aHi, b3, acc[7]);
    if (more) {
      *(v8bf*)(BsNext + boff) = brn;
      __syncthreads();
    }
  };

  for (int s2 = 0; s2 < KSTEPS / 2; ++s2) {
    step(2 * s2,     &Bs[0][0], &Bs[1][0], arA, arB);
    step(2 * s2 + 1, &Bs[1][0], &Bs[0][0], arB, arA);
  }

  const float bias[4] = { bkv[n0 + lmod], bkv[n0 + 16 + lmod], bkv[n0 + 32 + lmod], bkv[n0 + 48 + lmod] };
#pragma unroll
  for (int hi = 0; hi < 2; ++hi)
#pragma unroll
  for (int r = 0; r < 8; ++r) {
    const int m = m0 + 16 * hi + r + 8 * lhalf;
    if (m >= KVROWS) continue;
    const int tt = m >> 5, bb = m & 31;
    const int kvidx = (tt < M_ + R_) ? tt : tt + L_;   // left-context inserted at [72,200)
#pragma unroll
    for (int j = 0; j < 4; ++j) {
      const int n = n0 + 16 * j + lmod;
      float v = acc[4 * hi + j][r] + bias[j];
      if (n < D_) {           // key half
        const int h = n >> 7, dd = n & 127;
        kbf[((size_t)(bb * H_ + h) * KV_ + kvidx) * DH + dd] = (__bf16)v;
        if (tt >= M_ + R_)
          out_key[(((size_t)(tt - (M_ + R_) + L_)) * B_ + bb) * D_ + n] = v;
      } else {                // value half (transposed per head)
        const int nv = n - D_;
        const int h = nv >> 7, dd = nv & 127;
        vT[((size_t)(bb * H_ + h) * DH + dd) * VSTRIDE + kvidx] = (__bf16)v;
        if (tt >= M_ + R_)
          out_val[(((size_t)(tt - (M_ + R_) + L_)) * B_ + bb) * D_ + nv] = v;
      }
    }
  }
}

// ---------------- insert left-context K/V (bf16 buffers + fp32 outputs)
__global__ void pack_lc_kernel(const float* __restrict__ lck, const float* __restrict__ lcv,
                               __bf16* __restrict__ kbf, __bf16* __restrict__ vT,
                               float* __restrict__ out_key, float* __restrict__ out_val)
{
  const int i = blockIdx.x * 256 + threadIdx.x;
  if (i >= L_ * B_ * D_) return;
  const int n = i & (D_ - 1);
  const int rb = i >> 10;
  const int b = rb & 31, s = rb >> 5;
  const float kv = lck[i], vv = lcv[i];
  const int h = n >> 7, dd = n & 127;
  const int kvidx = M_ + R_ + s;
  kbf[((size_t)(b * H_ + h) * KV_ + kvidx) * DH + dd] = (__bf16)kv;
  vT [((size_t)(b * H_ + h) * DH + dd) * VSTRIDE + kvidx] = (__bf16)vv;
  out_key[i] = kv;
  out_val[i] = vv;
}

// ---------------- flash attention: per wave 16 T-rows x 128 dh, online softmax over 712 kv
__global__ __launch_bounds__(128) void flash_kernel(
    const __bf16* __restrict__ qbf, const __bf16* __restrict__ kbf, const __bf16* __restrict__ vT,
    const int* __restrict__ p_mkv, const int* __restrict__ p_mm,
    __bf16* __restrict__ attn)
{
  __shared__ __attribute__((aligned(32))) __bf16 smem[4][16 * 32];
  const int lane = threadIdx.x & 31, wave = threadIdx.x >> 5;
  const int lmod = lane & 15, lhalf = lane >> 4;
  const int bh = blockIdx.x;
  const int t0 = blockIdx.y * 64 + wave * 16;
  if (t0 >= T_) return;

  const int mkv = p_mkv[0], mm = p_mm[0];
  const int mask_lo = M_ - mm;
  const int mask_hi = M_ + R_ + L_ - mkv;

  v16bf qf[4];
  {
    int tq = t0 + lmod; if (tq >= T_) tq = T_ - 1;
    const __bf16* qrow = qbf + ((size_t)bh * T_ + tq) * DH;
#pragma unroll
    for (int dk = 0; dk < 4; ++dk) {
      v8bf lo = *(const v8bf*)(qrow + dk * 32 + lhalf * 16);
      v8bf hi = *(const v8bf*)(qrow + dk * 32 + lhalf * 16 + 8);
      qf[dk] = cat8(lo, hi);
    }
  }

  auto load_ktile = [&](int kc, v16bf* kb) {
    const int k0 = kc * 32;
    int kc0 = k0 + lmod;      if (kc0 >= KV_) kc0 = KV_ - 1;
    int kc1 = k0 + 16 + lmod; if (kc1 >= KV_) kc1 = KV_ - 1;
    const __bf16* kr0 = kbf + ((size_t)bh * KV_ + kc0) * DH + lhalf * 16;
    const __bf16* kr1 = kbf + ((size_t)bh * KV_ + kc1) * DH + lhalf * 16;
#pragma unroll
    for (int dk = 0; dk < 4; ++dk) {
      kb[dk]     = *(const v16bf*)(kr0 + dk * 32);
      kb[4 + dk] = *(const v16bf*)(kr1 + dk * 32);
    }
  };

  v8f zacc = {};
  v8f o[8];
#pragma unroll
  for (int j = 0; j < 8; ++j) o[j] = zacc;
  float mst[8], lst[8];
#pragma unroll
  for (int r = 0; r < 8; ++r) { mst[r] = -1e30f; lst[r] = 0.f; }

  // one attention chunk of 32 kv columns; K tile ping-pongs between
  // register arrays (kbCur consumed, kbNext prefetched -> no copies)
  auto chunk = [&](int kc, v16bf* kbCur, v16bf* kbNext, bool prefetch) {
    const int k0 = kc * 32;
    v8f s0 = zacc, s1 = zacc;
#pragma unroll
    for (int dk = 0; dk < 4; ++dk) {
      s0 = WMMA_BF16(qf[dk], kbCur[dk], s0);
      s1 = WMMA_BF16(qf[dk], kbCur[4 + dk], s1);
    }
    // issue V (current) + K (next) loads; softmax VALU below hides them
    v16bf vb[8];
#pragma unroll
    for (int j = 0; j < 8; ++j)
      vb[j] = *(const v16bf*)(vT + ((size_t)bh * DH + j * 16 + lmod) * VSTRIDE + k0 + lhalf * 16);
    if (prefetch) load_ktile(kc + 1, kbNext);

    float pr0[8], pr1[8], corr[8];
#pragma unroll
    for (int r = 0; r < 8; ++r) {
      const int t  = t0 + 8 * lhalf + r;
      const int ka = k0 + lmod, kb_ = k0 + 16 + lmod;
      float sa = s0[r], sb = s1[r];
      if (ka  < mask_lo || (ka  >= M_ + R_ && ka  < mask_hi) || (t == T_ - 1 && ka  < M_) || ka  >= KV_) sa = NEG_INF_;
      if (kb_ < mask_lo || (kb_ >= M_ + R_ && kb_ < mask_hi) || (t == T_ - 1 && kb_ < M_) || kb_ >= KV_) sb = NEG_INF_;
      float mx = fmaxf(sa, sb);
#pragma unroll
      for (int off = 1; off < 16; off <<= 1) mx = fmaxf(mx, __shfl_xor(mx, off, 16));
      const float mn = fmaxf(mst[r], mx);
      const float c  = __expf(mst[r] - mn);
      mst[r] = mn;
      const float pa = __expf(sa - mn), pb = __expf(sb - mn);
      float rs = pa + pb;
#pragma unroll
      for (int off = 1; off < 16; off <<= 1) rs += __shfl_xor(rs, off, 16);
      lst[r] = lst[r] * c + rs;
      corr[r] = c; pr0[r] = pa; pr1[r] = pb;
    }
#pragma unroll
    for (int j = 0; j < 8; ++j)
#pragma unroll
      for (int r = 0; r < 8; ++r) o[j][r] *= corr[r];

#pragma unroll
    for (int r = 0; r < 8; ++r) {
      smem[wave][(r + 8 * lhalf) * 32 + lmod]      = (__bf16)pr0[r];
      smem[wave][(r + 8 * lhalf) * 32 + 16 + lmod] = (__bf16)pr1[r];
    }
    asm volatile("s_wait_dscnt 0" ::: "memory");
    v16bf pf;
    {
      const __bf16* prow = &smem[wave][lmod * 32];
      v8bf lo = *(const v8bf*)(prow + lhalf * 16);
      v8bf hi = *(const v8bf*)(prow + lhalf * 16 + 8);
      pf = cat8(lo, hi);
    }
#pragma unroll
    for (int j = 0; j < 8; ++j)
      o[j] = WMMA_BF16(pf, vb[j], o[j]);
  };

  v16bf kbA[8], kbB[8];
  load_ktile(0, kbA);
  for (int kc2 = 0; kc2 < 11; ++kc2) {
    chunk(2 * kc2,     kbA, kbB, true);
    chunk(2 * kc2 + 1, kbB, kbA, true);
  }
  chunk(22, kbA, kbB, false);

  const int b = bh >> 3, h = bh & 7;
#pragma unroll
  for (int r = 0; r < 8; ++r) {
    const int t = t0 + 8 * lhalf + r;
    if (t >= T_) continue;
    const float inv = lst[r] > 0.f ? 1.0f / lst[r] : 0.f;
    const size_t rowbase = ((size_t)t * B_ + b) * D_ + h * DH;
#pragma unroll
    for (int j = 0; j < 8; ++j)
      attn[rowbase + j * 16 + lmod] = (__bf16)(o[j][r] * inv);
  }
}

// ---------------- output projection: out = attn @ Wo^T + bo, split rc / clipped mems
__global__ __launch_bounds__(256) void gemm_out_kernel(
    const __bf16* __restrict__ attn, const __bf16* __restrict__ Wo, const float* __restrict__ bo,
    float* __restrict__ out_rc, float* __restrict__ out_mems)
{
  __shared__ __attribute__((aligned(32))) __bf16 Bs[2][64 * 32];
  const int tid  = threadIdx.x;
  const int lane = tid & 31, wave = tid >> 5;
  const int lmod = lane & 15, lhalf = lane >> 4;
  const int m0 = blockIdx.y * 256 + wave * 32;
  const int n0 = blockIdx.x * 64;

  const int brow = tid >> 2, bq4 = tid & 3;
  const __bf16* bsrc = Wo + (size_t)(n0 + brow) * D_ + bq4 * 8;
  const int boff = brow * 32 + ((bq4 & 1) * 2 + (bq4 >> 1)) * 8;

  int arow0 = m0 + lmod;      if (arow0 >= QROWS) arow0 = QROWS - 1;
  int arow1 = m0 + 16 + lmod; if (arow1 >= QROWS) arow1 = QROWS - 1;
  const __bf16* arp0 = attn + (size_t)arow0 * D_ + lhalf * 8;
  const __bf16* arp1 = attn + (size_t)arow1 * D_ + lhalf * 8;

  *(v8bf*)(&Bs[0][boff]) = *(const v8bf*)(bsrc);
  v8bf arA[4], arB[4];
  arA[0] = *(const v8bf*)(arp0);
  arA[1] = *(const v8bf*)(arp0 + 16);
  arA[2] = *(const v8bf*)(arp1);
  arA[3] = *(const v8bf*)(arp1 + 16);
  __syncthreads();

  v8f acc[8];
#pragma unroll
  for (int j = 0; j < 8; ++j) acc[j] = v8f{};

  auto step = [&](int s, __bf16* BsCur, __bf16* BsNext, v8bf* arCur, v8bf* arNext) {
    const bool more = (s + 1) < KSTEPS;
    v8bf brn{};
    if (more) {
      const int kn = (s + 1) * 32;
      brn = *(const v8bf*)(bsrc + kn);
      arNext[0] = *(const v8bf*)(arp0 + kn);
      arNext[1] = *(const v8bf*)(arp0 + kn + 16);
      arNext[2] = *(const v8bf*)(arp1 + kn);
      arNext[3] = *(const v8bf*)(arp1 + kn + 16);
    }
    v16bf aLo = cat8(arCur[0], arCur[1]);
    v16bf aHi = cat8(arCur[2], arCur[3]);
    const __bf16* bb = BsCur + lmod * 32 + lhalf * 16;
    v16bf b0 = *(const v16bf*)(bb +  0 * 32);
    v16bf b1 = *(const v16bf*)(bb + 16 * 32);
    v16bf b2 = *(const v16bf*)(bb + 32 * 32);
    v16bf b3 = *(const v16bf*)(bb + 48 * 32);
    acc[0] = WMMA_BF16(aLo, b0, acc[0]);
    acc[1] = WMMA_BF16(aLo, b1, acc[1]);
    acc[2] = WMMA_BF16(aLo, b2, acc[2]);
    acc[3] = WMMA_BF16(aLo, b3, acc[3]);
    acc[4] = WMMA_BF16(aHi, b0, acc[4]);
    acc[5] = WMMA_BF16(aHi, b1, acc[5]);
    acc[6] = WMMA_BF16(aHi, b2, acc[6]);
    acc[7] = WMMA_BF16(aHi, b3, acc[7]);
    if (more) {
      *(v8bf*)(BsNext + boff) = brn;
      __syncthreads();
    }
  };

  for (int s2 = 0; s2 < KSTEPS / 2; ++s2) {
    step(2 * s2,     &Bs[0][0], &Bs[1][0], arA, arB);
    step(2 * s2 + 1, &Bs[1][0], &Bs[0][0], arB, arA);
  }

  const float bias[4] = { bo[n0 + lmod], bo[n0 + 16 + lmod], bo[n0 + 32 + lmod], bo[n0 + 48 + lmod] };
#pragma unroll
  for (int hi = 0; hi < 2; ++hi)
#pragma unroll
  for (int r = 0; r < 8; ++r) {
    const int m = m0 + 16 * hi + r + 8 * lhalf;
    if (m >= QROWS) continue;
    const int tt = m >> 5, bb = m & 31;
#pragma unroll
    for (int j = 0; j < 4; ++j) {
      const int n = n0 + 16 * j + lmod;
      float v = acc[4 * hi + j][r] + bias[j];
      if (tt < T_ - S_) {
        out_rc[((size_t)tt * B_ + bb) * D_ + n] = v;
      } else {
        out_mems[(size_t)bb * D_ + n] = fminf(10.f, fmaxf(-10.f, v));
      }
    }
  }
}

// ---------------- host launcher ----------------
extern "C" void kernel_launch(void* const* d_in, const int* in_sizes, int n_in,
                              void* d_out, int out_size, void* d_ws, size_t ws_size,
                              hipStream_t stream) {
  (void)in_sizes; (void)n_in; (void)out_size; (void)ws_size;
  const float* utt  = (const float*)d_in[0];
  const float* rc   = (const float*)d_in[1];
  const float* summ = (const float*)d_in[2];
  const float* mems = (const float*)d_in[3];
  const float* lck  = (const float*)d_in[4];
  const float* lcv  = (const float*)d_in[5];
  const float* Wq   = (const float*)d_in[6];
  const float* bq   = (const float*)d_in[7];
  const float* Wkv  = (const float*)d_in[8];
  const float* bkv  = (const float*)d_in[9];
  const float* Wo   = (const float*)d_in[10];
  const float* bo   = (const float*)d_in[11];
  const int*   mkv  = (const int*)d_in[12];
  const int*   mm   = (const int*)d_in[13];

  float* out       = (float*)d_out;
  float* out_rc    = out;                                      // (576,32,1024)
  float* out_mems  = out + (size_t)(T_ - S_) * B_ * D_;        // (1,32,1024)
  float* out_key   = out + (size_t)T_ * B_ * D_;               // (640,32,1024)
  float* out_val   = out_key + (size_t)(L_ + U_) * B_ * D_;    // (640,32,1024)

  char* ws = (char*)d_ws;
  __bf16* Wq_bf  = (__bf16*)ws; ws += (size_t)D_ * D_ * 2;
  __bf16* Wkv_bf = (__bf16*)ws; ws += (size_t)2 * D_ * D_ * 2;
  __bf16* Wo_bf  = (__bf16*)ws; ws += (size_t)D_ * D_ * 2;
  __bf16* q_bf   = (__bf16*)ws; ws += (size_t)B_ * H_ * T_ * DH * 2;
  __bf16* k_bf   = (__bf16*)ws; ws += (size_t)B_ * H_ * KV_ * DH * 2;
  __bf16* vT_bf  = (__bf16*)ws; ws += (size_t)B_ * H_ * DH * VSTRIDE * 2;
  __bf16* attn_bf= (__bf16*)ws;

  // 1) weights -> bf16
  cvt_f32_bf16<<<(D_ * D_ + 255) / 256, 256, 0, stream>>>(Wq, Wq_bf, D_ * D_);
  cvt_f32_bf16<<<(2 * D_ * D_ + 255) / 256, 256, 0, stream>>>(Wkv, Wkv_bf, 2 * D_ * D_);
  cvt_f32_bf16<<<(D_ * D_ + 255) / 256, 256, 0, stream>>>(Wo, Wo_bf, D_ * D_);

  // 2) Q projection (fused concat gather + scale)
  gemm_q_kernel<<<dim3(D_ / 64, (QROWS + 255) / 256), 256, 0, stream>>>(
      utt, rc, summ, Wq_bf, bq, q_bf);

  // 3) KV projection (fused gather; bf16 K / transposed V + fp32 key/value outputs)
  gemm_kv_kernel<<<dim3(2 * D_ / 64, (KVROWS + 255) / 256), 256, 0, stream>>>(
      mems, rc, utt, Wkv_bf, bkv, k_bf, vT_bf, out_key, out_val);

  // 4) splice left-context K/V
  pack_lc_kernel<<<(L_ * B_ * D_ + 255) / 256, 256, 0, stream>>>(
      lck, lcv, k_bf, vT_bf, out_key, out_val);

  // 5) flash attention (256 bh x 10 T-tiles, 4 waves/block)
  flash_kernel<<<dim3(B_ * H_, (T_ + 63) / 64), 128, 0, stream>>>(
      q_bf, k_bf, vT_bf, mkv, mm, attn_bf);

  // 6) output projection + clip/split
  gemm_out_kernel<<<dim3(D_ / 64, (QROWS + 255) / 256), 256, 0, stream>>>(
      attn_bf, Wo_bf, bo, out_rc, out_mems);
}
